// SASRec_52536039965142
// MI455X (gfx1250) — matrix operand order repaired and checked by reading, not compiled, not run on hardware
//
#include <hip/hip_runtime.h>

// ---------------------------------------------------------------------------
// SASRec forward for MI455X (gfx1250, wave32, WMMA).
// All big matmuls use v_wmma_f32_16x16x32_bf16 (bf16 A/B, f32 accumulate).
// Roofline: logits GEMM output (384 MB f32) is the memory floor (~16.5us @
// 23.3 TB/s); bf16 WMMA keeps its 49 GFLOP under that floor, and the logits
// output is stored with the NT temporal hint so the 15 MB bf16 item_emb B
// operand stays resident in the 192 MB L2 across the whole GEMM.
// ---------------------------------------------------------------------------

typedef unsigned short u16;
typedef __attribute__((ext_vector_type(16))) __bf16 v16bf;
typedef __attribute__((ext_vector_type(8)))  float  v8f;

#define Bdim 16
#define Ldim 200
#define Ddim 256
#define Hdim 4
#define HD   64
#define FFN  1024
#define Vp1  30001
#define Vpad 30016      // pad N to multiple of 64 for logits tiles
#define Lpad 224        // pad key dim to multiple of 32 for attn@V
#define MTOK (Bdim*Ldim) // 3200 tokens, multiple of 16

__device__ __forceinline__ u16 f2bf(float f) {
  unsigned u = __float_as_uint(f);
  u += 0x7FFFu + ((u >> 16) & 1u);        // round-to-nearest-even
  return (u16)(u >> 16);
}

union FragU { v16bf v; uint4 q[2]; };

// 16-bit A-matrix 16x32 layout (ISA 7.12.2): lane l (m=l%16, hi=l/16) holds
// row m, K = k0 + hi*8 + {0..7} in V0..3 and K = k0 + 16 + hi*8 + {0..7} in V4..7.
__device__ __forceinline__ v16bf frag_a(const u16* rowptr, int k0, int hi) {
  FragU f;
  const u16* p = rowptr + k0 + hi * 8;
  f.q[0] = *(const uint4*)(p);
  f.q[1] = *(const uint4*)(p + 16);
  return f.v;
}

// 16-bit B-matrix 32x16 layout (same K order as A):
// lane l (n=l%16, hi=l/16) holds column n, K = k0 + hi*8 + {0..7} and k0 + 16 + hi*8 + {0..7}.
// colptr = &Bt[n * ldb]  (Bt stored N x K row-major).
__device__ __forceinline__ v16bf frag_b(const u16* colptr, int k0, int hi) {
  FragU f;
  const u16* p = colptr + k0 + hi * 8;
  f.q[0] = *(const uint4*)(p);
  f.q[1] = *(const uint4*)(p + 16);
  return f.v;
}

__device__ __forceinline__ v8f wmma_bf16(v16bf a, v16bf b, v8f c) {
  return __builtin_amdgcn_wmma_f32_16x16x32_bf16(false, a, false, b, (short)0, c,
                                                 false, false);
}

// ---------------------------------------------------------------------------
// Embedding: x = (emb[id]*sqrt(D) + pos) * mask ; also bf16 copy + row mask
// ---------------------------------------------------------------------------
__global__ void k_embed(const int* __restrict__ ids, const float* __restrict__ emb,
                        const float* __restrict__ pos, float* __restrict__ x,
                        u16* __restrict__ xb, float* __restrict__ rmask) {
  int m = blockIdx.x, d = threadIdx.x;
  int id = ids[m];
  int l = m % Ldim;
  float v = (id != 0) ? (emb[(size_t)id * Ddim + d] * 16.0f + pos[l * Ddim + d]) : 0.0f;
  size_t idx = (size_t)m * Ddim + d;
  x[idx] = v;
  xb[idx] = f2bf(v);
  if (d == 0) rmask[m] = (id != 0) ? 1.0f : 0.0f;
}

// Elementwise f32 -> bf16
__global__ void k_cvt(const float* __restrict__ s, u16* __restrict__ d, int n) {
  int i = blockIdx.x * 256 + threadIdx.x;
  if (i < n) d[i] = f2bf(s[i]);
}

// Transpose-convert weight (K x N f32, row-major) -> (N x K bf16, row-major)
__global__ void k_cvtT(const float* __restrict__ w, u16* __restrict__ wt, int K, int N) {
  int i = blockIdx.x * 256 + threadIdx.x;
  if (i >= K * N) return;
  int k = i / N, n = i - k * N;
  wt[(size_t)n * K + k] = f2bf(w[i]);
}

// ---------------------------------------------------------------------------
// LayerNorm over D=256.  val = A + (Badd?).  Optionally store pre-LN sum,
// LN result f32, LN result bf16.
// ---------------------------------------------------------------------------
__global__ void k_ln(const float* __restrict__ A, const float* __restrict__ Badd,
                     const float* __restrict__ g, const float* __restrict__ be,
                     float* __restrict__ sum_out, float* __restrict__ lnf,
                     u16* __restrict__ lnb) {
  __shared__ float red[Ddim];
  int m = blockIdx.x, d = threadIdx.x;
  size_t idx = (size_t)m * Ddim + d;
  float v = A[idx] + (Badd ? Badd[idx] : 0.0f);
  if (sum_out) sum_out[idx] = v;
  red[d] = v;
  __syncthreads();
  for (int s = 128; s > 0; s >>= 1) { if (d < s) red[d] += red[d + s]; __syncthreads(); }
  float mu = red[0] * (1.0f / Ddim);
  __syncthreads();
  float dv = v - mu;
  red[d] = dv * dv;
  __syncthreads();
  for (int s = 128; s > 0; s >>= 1) { if (d < s) red[d] += red[d + s]; __syncthreads(); }
  float var = red[0] * (1.0f / Ddim);
  float o = dv * rsqrtf(var + 1e-8f) * g[d] + be[d];
  if (lnf) lnf[idx] = o;
  if (lnb) lnb[idx] = f2bf(o);
}

// ---------------------------------------------------------------------------
// Generic NT GEMM on WMMA: C[M,N] = A[M,K] @ Bt[N,K]^T (+bias)(relu)(+resid)(*rmask)
// One wave per 16x64 output strip (A fragment reused across 4 N-tiles).
// Optional outputs: f32 C (optionally with NT store hint), bf16 C, and "vt"
// scatter (V transposed per head for the attention B operand:
// vt[((b*H+h)*HD+d)*Lpad + t]).
// ---------------------------------------------------------------------------
__global__ void k_gemm(const u16* __restrict__ A, const u16* __restrict__ Bt,
                       const float* __restrict__ bias, float* __restrict__ Cf,
                       u16* __restrict__ Cb, const float* __restrict__ resid,
                       const float* __restrict__ rmask, u16* __restrict__ vt,
                       int M, int N, int K, int relu, int ntstore) {
  int lane = threadIdx.x, l16 = lane & 15, hi = lane >> 4;
  int n0 = blockIdx.x * 64, m0 = blockIdx.y * 16;
  const u16* Arow = A + (size_t)(m0 + l16) * K;
  v8f acc[4] = {{}, {}, {}, {}};
#pragma unroll 4
  for (int k0 = 0; k0 < K; k0 += 32) {
    v16bf a = frag_a(Arow, k0, hi);
#pragma unroll
    for (int j = 0; j < 4; ++j) {
      const u16* Bcol = Bt + (size_t)(n0 + j * 16 + l16) * K;
      acc[j] = wmma_bf16(a, frag_b(Bcol, k0, hi), acc[j]);
    }
  }
#pragma unroll
  for (int j = 0; j < 4; ++j) {
    int n = n0 + j * 16 + l16;
    float bn = (bias && n < N) ? bias[n] : 0.0f;
#pragma unroll
    for (int v = 0; v < 8; ++v) {
      int m = m0 + hi * 8 + v;
      float val = acc[j][v] + bn;
      if (relu) val = fmaxf(val, 0.0f);
      if (n < N) {
        size_t idx = (size_t)m * N + n;
        if (resid) val += resid[idx];
        if (rmask) val *= rmask[m];
        if (Cf) {
          if (ntstore) __builtin_nontemporal_store(val, &Cf[idx]);
          else Cf[idx] = val;
        }
        if (Cb) Cb[idx] = f2bf(val);
        if (vt) {
          int bb = m / Ldim, t = m - bb * Ldim;
          int hh = n >> 6, dd = n & 63;
          vt[(((size_t)(bb * Hdim + hh) * HD) + dd) * Lpad + t] = f2bf(val);
        }
      }
    }
  }
}

// ---------------------------------------------------------------------------
// Attention: one wave per (query-16-tile, head, batch).
// scores = Q K^T * scale  (WMMA, K-dim = HD = 64)
// masked softmax in LDS (f32), probs -> bf16 (zero-padded keys to Lpad)
// out = P @ V            (WMMA, K-dim = Lpad = 224, V pre-transposed)
// ---------------------------------------------------------------------------
__global__ void k_attn(const u16* __restrict__ Qb, const u16* __restrict__ Kb,
                       const u16* __restrict__ Vtb, const float* __restrict__ rmask,
                       float* __restrict__ aout) {
  __shared__ __align__(16) float sc[16][Lpad];
  __shared__ __align__(16) u16   pb[16][Lpad];
  int lane = threadIdx.x, l16 = lane & 15, hi = lane >> 4;
  int mt = blockIdx.x;   // 0..12 (13 query tiles)
  int h  = blockIdx.y;   // 0..3
  int b  = blockIdx.z;   // 0..15

  int q  = mt * 16 + l16;
  int qc = q < Ldim ? q : (Ldim - 1);
  const u16* Qrow = Qb + ((size_t)(b * Ldim + qc) * Ddim) + h * HD;
  v16bf a0 = frag_a(Qrow, 0, hi);   // head dims 0..31
  v16bf a1 = frag_a(Qrow, 32, hi);  // head dims 32..63

  // ---- scores: 13 key tiles of 16 ----
  for (int nt = 0; nt < 13; ++nt) {
    int key = nt * 16 + l16;
    int kc  = key < Ldim ? key : (Ldim - 1);
    const u16* Krow = Kb + ((size_t)(b * Ldim + kc) * Ddim) + h * HD;
    v8f c = {};
    c = wmma_bf16(a0, frag_b(Krow, 0, hi), c);
    c = wmma_bf16(a1, frag_b(Krow, 32, hi), c);
#pragma unroll
    for (int v = 0; v < 8; ++v)
      sc[hi * 8 + v][nt * 16 + l16] = c[v] * 0.125f;  // 1/sqrt(HD)
  }
  __syncthreads();

  // ---- masked softmax, row by row (wave-cooperative over 224 cols) ----
  int tokbase = b * Ldim;
  for (int r = 0; r < 16; ++r) {
    int qi = mt * 16 + r;
    float vals[7];
    float mx = -3.4e38f;
#pragma unroll
    for (int s = 0; s < 7; ++s) {
      int k = s * 32 + lane;
      float vv = sc[r][k];
      bool bad = (k > qi) || (k >= Ldim) || (rmask[tokbase + (k < Ldim ? k : 0)] == 0.0f);
      vv = bad ? -1.0e9f : vv;
      vals[s] = vv;
      mx = fmaxf(mx, vv);
    }
    for (int o = 16; o > 0; o >>= 1) mx = fmaxf(mx, __shfl_xor(mx, o, 32));
    float sum = 0.0f;
    float es[7];
#pragma unroll
    for (int s = 0; s < 7; ++s) { float e = __expf(vals[s] - mx); es[s] = e; sum += e; }
    for (int o = 16; o > 0; o >>= 1) sum += __shfl_xor(sum, o, 32);
    float qm  = (qi < Ldim) ? rmask[tokbase + qi] : 0.0f;
    float inv = qm / sum;
#pragma unroll
    for (int s = 0; s < 7; ++s) {
      int k = s * 32 + lane;
      float p = (k < Ldim) ? es[s] * inv : 0.0f;
      pb[r][k] = f2bf(p);
    }
  }
  __syncthreads();

  // ---- out = P @ V ----
  const u16* Vbase = Vtb + (size_t)(b * Hdim + h) * HD * Lpad;
  const u16* Prow  = &pb[l16][0];
#pragma unroll
  for (int dt = 0; dt < 4; ++dt) {
    const u16* Vcol = Vbase + (size_t)(dt * 16 + l16) * Lpad;
    v8f c = {};
    for (int kc = 0; kc < 7; ++kc)
      c = wmma_bf16(frag_a(Prow, kc * 32, hi), frag_b(Vcol, kc * 32, hi), c);
#pragma unroll
    for (int v = 0; v < 8; ++v) {
      int qi = mt * 16 + hi * 8 + v;
      if (qi < Ldim)
        aout[((size_t)(b * Ldim + qi)) * Ddim + h * HD + dt * 16 + l16] = c[v];
    }
  }
}

// ---------------------------------------------------------------------------
// Host-side orchestration
// ---------------------------------------------------------------------------
extern "C" void kernel_launch(void* const* d_in, const int* in_sizes, int n_in,
                              void* d_out, int out_size, void* d_ws, size_t ws_size,
                              hipStream_t stream) {
  (void)in_sizes; (void)n_in; (void)out_size; (void)ws_size;
  const int*   ids      = (const int*)d_in[0];
  const float* item_emb = (const float*)d_in[1];
  const float* pos_emb  = (const float*)d_in[2];
  // blocks: Wq,bq,Wk,bk,Wv,bv,W1,b1,W2,b2,g1,be1,g2,be2 (dict order), x2
  const float* gF = (const float*)d_in[3 + 2 * 14 + 0];
  const float* bF = (const float*)d_in[3 + 2 * 14 + 1];
  float* logits = (float*)d_out;

  // workspace carve-up (256B aligned)
  char* w = (char*)d_ws;
  auto alloc = [&](size_t bytes) { void* p = (void*)w; w += (bytes + 255) & ~(size_t)255; return p; };
  float* x     = (float*)alloc((size_t)MTOK * Ddim * 4);
  float* qin   = (float*)alloc((size_t)MTOK * Ddim * 4);
  float* ores  = (float*)alloc((size_t)MTOK * Ddim * 4);
  float* aout  = (float*)alloc((size_t)MTOK * Ddim * 4);
  float* rmask = (float*)alloc((size_t)MTOK * 4);
  u16* xb   = (u16*)alloc((size_t)MTOK * Ddim * 2);
  u16* qinb = (u16*)alloc((size_t)MTOK * Ddim * 2);
  u16* Qb   = (u16*)alloc((size_t)MTOK * Ddim * 2);
  u16* Kb   = (u16*)alloc((size_t)MTOK * Ddim * 2);
  u16* hb   = (u16*)alloc((size_t)MTOK * Ddim * 2);
  u16* xfb  = (u16*)alloc((size_t)MTOK * Ddim * 2);
  u16* y1b  = (u16*)alloc((size_t)MTOK * FFN * 2);
  u16* Vtb  = (u16*)alloc((size_t)Bdim * Hdim * HD * Lpad * 2);
  u16* embb = (u16*)alloc((size_t)Vpad * Ddim * 2);
  u16* Wqt  = (u16*)alloc((size_t)Ddim * Ddim * 2);
  u16* Wkt  = (u16*)alloc((size_t)Ddim * Ddim * 2);
  u16* Wvt  = (u16*)alloc((size_t)Ddim * Ddim * 2);
  u16* W1t  = (u16*)alloc((size_t)Ddim * FFN * 2);
  u16* W2t  = (u16*)alloc((size_t)FFN * Ddim * 2);

  // 1) embedding + mask
  k_embed<<<MTOK, Ddim, 0, stream>>>(ids, item_emb, pos_emb, x, xb, rmask);
  // 2) item_emb -> bf16 (B operand of logits GEMM; already N x K layout)
  k_cvt<<<(Vp1 * Ddim + 255) / 256, 256, 0, stream>>>(item_emb, embb, Vp1 * Ddim);

  for (int blk = 0; blk < 2; ++blk) {
    int base = 3 + blk * 14;
    const float* Wq = (const float*)d_in[base + 0];
    const float* bq = (const float*)d_in[base + 1];
    const float* Wk = (const float*)d_in[base + 2];
    const float* bk = (const float*)d_in[base + 3];
    const float* Wv = (const float*)d_in[base + 4];
    const float* bv = (const float*)d_in[base + 5];
    const float* W1 = (const float*)d_in[base + 6];
    const float* b1 = (const float*)d_in[base + 7];
    const float* W2 = (const float*)d_in[base + 8];
    const float* b2 = (const float*)d_in[base + 9];
    const float* g1 = (const float*)d_in[base + 10];
    const float* be1= (const float*)d_in[base + 11];
    const float* g2 = (const float*)d_in[base + 12];
    const float* be2= (const float*)d_in[base + 13];

    // weights -> bf16 transposed (N x K)
    k_cvtT<<<(Ddim * Ddim + 255) / 256, 256, 0, stream>>>(Wq, Wqt, Ddim, Ddim);
    k_cvtT<<<(Ddim * Ddim + 255) / 256, 256, 0, stream>>>(Wk, Wkt, Ddim, Ddim);
    k_cvtT<<<(Ddim * Ddim + 255) / 256, 256, 0, stream>>>(Wv, Wvt, Ddim, Ddim);
    k_cvtT<<<(Ddim * FFN  + 255) / 256, 256, 0, stream>>>(W1, W1t, Ddim, FFN);
    k_cvtT<<<(FFN  * Ddim + 255) / 256, 256, 0, stream>>>(W2, W2t, FFN, Ddim);

    // LN1: q_in = LN(x)
    k_ln<<<MTOK, Ddim, 0, stream>>>(x, nullptr, g1, be1, nullptr, qin, qinb);

    dim3 gQKV(Ddim / 64, MTOK / 16);
    // Q = q_in@Wq + bq  (bf16 out only: A of scores)
    k_gemm<<<gQKV, 32, 0, stream>>>(qinb, Wqt, bq, nullptr, Qb, nullptr, nullptr,
                                    nullptr, MTOK, Ddim, Ddim, 0, 0);
    // K = x@Wk + bk     (bf16 out only: B^T of scores, row-major token x dim)
    k_gemm<<<gQKV, 32, 0, stream>>>(xb, Wkt, bk, nullptr, Kb, nullptr, nullptr,
                                    nullptr, MTOK, Ddim, Ddim, 0, 0);
    // V = x@Wv + bv     (scatter transposed per head -> Vtb)
    k_gemm<<<gQKV, 32, 0, stream>>>(xb, Wvt, bv, nullptr, nullptr, nullptr, nullptr,
                                    Vtb, MTOK, Ddim, Ddim, 0, 0);

    // attention
    k_attn<<<dim3(13, Hdim, Bdim), 32, 0, stream>>>(Qb, Kb, Vtb, rmask, aout);

    // LN2: ores = aout + q_in ; h = LN(ores) (bf16)
    k_ln<<<MTOK, Ddim, 0, stream>>>(aout, qin, g2, be2, ores, nullptr, hb);

    // FFN1: y1 = relu(h@W1 + b1) (bf16)
    k_gemm<<<dim3(FFN / 64, MTOK / 16), 32, 0, stream>>>(hb, W1t, b1, nullptr, y1b,
                                                         nullptr, nullptr, nullptr,
                                                         MTOK, FFN, Ddim, 1, 0);
    // FFN2: x = (y1@W2 + b2 + ores) * mask  (f32 + bf16)
    k_gemm<<<dim3(Ddim / 64, MTOK / 16), 32, 0, stream>>>(y1b, W2t, b2, x, xb, ores,
                                                          rmask, nullptr,
                                                          MTOK, Ddim, FFN, 0, 0);
  }

  // final LN -> bf16
  k_ln<<<MTOK, Ddim, 0, stream>>>(x, nullptr, gF, bF, nullptr, nullptr, xfb);

  // logits = xf @ item_emb^T   (M=3200, N=30001, K=256), write-bandwidth bound;
  // NT stores keep the streamed 384 MB output from evicting emb_b out of L2.
  k_gemm<<<dim3(Vpad / 64, MTOK / 16), 32, 0, stream>>>(xfb, embb, nullptr, logits,
                                                        nullptr, nullptr, nullptr,
                                                        nullptr, MTOK, Vp1, Ddim, 0, 1);
}
